// QuantizedLinear_30313879175844
// MI455X (gfx1250) — compile-verified, matrix operand order checked by reading, not run on hardware
//
#include <hip/hip_runtime.h>
#include <hip/hip_bf16.h>

typedef __attribute__((ext_vector_type(16))) _Float16 v16h;
typedef __attribute__((ext_vector_type(8)))  _Float16 v8h;
typedef __attribute__((ext_vector_type(8)))  float    v8f;

#define IN_FEATURES  4096
#define OUT_FEATURES 11008
#define TOKENS       8
#define WAVES        8
#define K_PER_WAVE   (IN_FEATURES / WAVES)   // 512
#define K_STEP       32                      // K of v_wmma_f32_16x16x32_f16

union V16 { v16h v; v8h h[2]; _Float16 e[16]; };

// ---------------------------------------------------------------------------
// Prep kernel 1: split x (fp32) into exact-sum hi/lo f16 pair, once per launch.
// ---------------------------------------------------------------------------
__global__ __launch_bounds__(256)
void prep_split_kernel(const float* __restrict__ x,
                       _Float16* __restrict__ xhi,
                       _Float16* __restrict__ xlo)
{
    const int i = blockIdx.x * 256 + threadIdx.x;   // grid covers 8*4096
    const float v = x[i];
    const _Float16 hi = (_Float16)v;
    xhi[i] = hi;
    xlo[i] = (_Float16)(v - (float)hi);
}

// ---------------------------------------------------------------------------
// Prep kernel 2: per-token row sums of x (for the zero-point correction term).
// ---------------------------------------------------------------------------
__global__ __launch_bounds__(256)
void prep_rowsum_kernel(const float* __restrict__ x, float* __restrict__ rowsum)
{
    __shared__ float sm[256];
    const int row = blockIdx.x;          // 0..7
    const int tid = threadIdx.x;
    float s = 0.0f;
#pragma unroll
    for (int i = 0; i < IN_FEATURES / 256; ++i)
        s += x[row * IN_FEATURES + i * 256 + tid];
    sm[tid] = s;
    __syncthreads();
    for (int off = 128; off > 0; off >>= 1) {
        if (tid < off) sm[tid] += sm[tid + off];
        __syncthreads();
    }
    if (tid == 0) rowsum[row] = sm[0];
}

// ---------------------------------------------------------------------------
// Main kernel (fast path): A pre-split in ws, zero A-side VALU in the loop.
// One block = one 16-wide output-channel tile; 8 waves split K.
// ---------------------------------------------------------------------------
__global__ __launch_bounds__(256)
void qlinear_wmma_fast(const _Float16* __restrict__ xhi,
                       const _Float16* __restrict__ xlo,
                       const int*   __restrict__ wq,
                       const float* __restrict__ scale,
                       const int*   __restrict__ zp,
                       const float* __restrict__ bias,
                       const float* __restrict__ rowsum,
                       float*       __restrict__ out)
{
    __shared__ float red[WAVES * 256];   // 8 KB: per-wave C tiles

    const int tid   = threadIdx.x;
    const int wave  = tid >> 5;
    const int lane  = tid & 31;
    const int obase = blockIdx.x * 16;

    // A (16x32 f16) layout: lanes 0-15 -> M=lane, K in {0..7,16..23};
    // lanes 16-31 -> M=lane-16, K in {8..15,24..31}. B (32x16) mirrors with N.
    const int mn = lane & 15;
    const int ko = (lane < 16) ? 0 : 8;

    // Padded token rows (m>=8) feed C rows we never read -> alias row (m&7),
    // which removes all divergence from the inner loop.
    const size_t arow = (size_t)(mn & (TOKENS - 1)) * IN_FEATURES;
    const _Float16* hirow = xhi + arow;
    const _Float16* lorow = xlo + arow;
    const int*      wrow  = wq + (size_t)(obase + mn) * IN_FEATURES;

    v8f acc = {};
    const int kwave = wave * K_PER_WAVE;

#pragma unroll 2
    for (int it = 0; it < K_PER_WAVE / K_STEP; ++it) {
        const int kb = kwave + it * K_STEP + ko;

        __builtin_prefetch(wrow + kb + 2 * K_STEP, 0, 1);   // weight stream

        // A: 4x b128 f16 loads, no conversion needed.
        V16 ah, al;
        ah.h[0] = *(const v8h*)(hirow + kb);
        ah.h[1] = *(const v8h*)(hirow + kb + 16);
        al.h[0] = *(const v8h*)(lorow + kb);
        al.h[1] = *(const v8h*)(lorow + kb + 16);

        // B: 4x b128 int32 loads (the 180MB stream), convert to f16 (exact).
        const int4 b0 = *(const int4*)(wrow + kb);
        const int4 b1 = *(const int4*)(wrow + kb + 4);
        const int4 b2 = *(const int4*)(wrow + kb + 16);
        const int4 b3 = *(const int4*)(wrow + kb + 20);
        const int bi[16] = { b0.x,b0.y,b0.z,b0.w, b1.x,b1.y,b1.z,b1.w,
                             b2.x,b2.y,b2.z,b2.w, b3.x,b3.y,b3.z,b3.w };
        V16 bh;
#pragma unroll
        for (int e = 0; e < 16; ++e)
            bh.e[e] = (_Float16)bi[e];   // |q| <= 128: exact in f16

        // Split-precision: x*q = x_hi*q + x_lo*q  (~22 mantissa bits)
        acc = __builtin_amdgcn_wmma_f32_16x16x32_f16(
                  false, ah.v, false, bh.v, (short)0, acc, false, false);
        acc = __builtin_amdgcn_wmma_f32_16x16x32_f16(
                  false, al.v, false, bh.v, (short)0, acc, false, false);
    }

#pragma unroll
    for (int r = 0; r < 8; ++r)
        red[wave * 256 + r * 32 + lane] = acc[r];
    __syncthreads();

    // C layout: VGPR r, lanes 0-15 -> (M=r, N=lane). TOKENS==8 rows all there.
    if (tid < TOKENS * 16) {
        const int m = tid >> 4;
        const int n = tid & 15;
        float s = 0.0f;
#pragma unroll
        for (int w = 0; w < WAVES; ++w)
            s += red[w * 256 + m * 32 + n];
        const int o = obase + n;
        // out = scale * (sum(x*q) - zp * sum(x)) + bias
        const float corr = (float)zp[o] * rowsum[m];
        out[(size_t)m * OUT_FEATURES + o] = (s - corr) * scale[o] + bias[o];
    }
}

// ---------------------------------------------------------------------------
// Fallback (no workspace): on-the-fly A split, zp folded into B in int domain.
// ---------------------------------------------------------------------------
__global__ __launch_bounds__(256)
void qlinear_wmma_fallback(const float* __restrict__ x,
                           const int*   __restrict__ wq,
                           const float* __restrict__ scale,
                           const int*   __restrict__ zp,
                           const float* __restrict__ bias,
                           float*       __restrict__ out)
{
    __shared__ float red[WAVES * 256];

    const int tid   = threadIdx.x;
    const int wave  = tid >> 5;
    const int lane  = tid & 31;
    const int obase = blockIdx.x * 16;
    const int mn = lane & 15;
    const int ko = (lane < 16) ? 0 : 8;

    const float* xrow = x + (size_t)(mn & (TOKENS - 1)) * IN_FEATURES;
    const int*   wrow = wq + (size_t)(obase + mn) * IN_FEATURES;
    const int    zpo  = zp[obase + mn];

    v8f acc = {};
    const int kwave = wave * K_PER_WAVE;

#pragma unroll 2
    for (int it = 0; it < K_PER_WAVE / K_STEP; ++it) {
        const int kb = kwave + it * K_STEP + ko;
        __builtin_prefetch(wrow + kb + 2 * K_STEP, 0, 1);

        const float4 a0 = *(const float4*)(xrow + kb);
        const float4 a1 = *(const float4*)(xrow + kb + 4);
        const float4 a2 = *(const float4*)(xrow + kb + 16);
        const float4 a3 = *(const float4*)(xrow + kb + 20);
        const int4 b0 = *(const int4*)(wrow + kb);
        const int4 b1 = *(const int4*)(wrow + kb + 4);
        const int4 b2 = *(const int4*)(wrow + kb + 16);
        const int4 b3 = *(const int4*)(wrow + kb + 20);

        const float af[16] = { a0.x,a0.y,a0.z,a0.w, a1.x,a1.y,a1.z,a1.w,
                               a2.x,a2.y,a2.z,a2.w, a3.x,a3.y,a3.z,a3.w };
        const int   bi[16] = { b0.x,b0.y,b0.z,b0.w, b1.x,b1.y,b1.z,b1.w,
                               b2.x,b2.y,b2.z,b2.w, b3.x,b3.y,b3.z,b3.w };
        V16 ah, al, bh;
#pragma unroll
        for (int e = 0; e < 16; ++e) {
            const _Float16 hi = (_Float16)af[e];
            ah.e[e] = hi;
            al.e[e] = (_Float16)(af[e] - (float)hi);
            bh.e[e] = (_Float16)(bi[e] - zpo);
        }
        acc = __builtin_amdgcn_wmma_f32_16x16x32_f16(
                  false, ah.v, false, bh.v, (short)0, acc, false, false);
        acc = __builtin_amdgcn_wmma_f32_16x16x32_f16(
                  false, al.v, false, bh.v, (short)0, acc, false, false);
    }

#pragma unroll
    for (int r = 0; r < 8; ++r)
        red[wave * 256 + r * 32 + lane] = acc[r];
    __syncthreads();

    if (tid < TOKENS * 16) {
        const int m = tid >> 4;
        const int n = tid & 15;
        float s = 0.0f;
#pragma unroll
        for (int w = 0; w < WAVES; ++w)
            s += red[w * 256 + m * 32 + n];
        const int o = obase + n;
        out[(size_t)m * OUT_FEATURES + o] = s * scale[o] + bias[o];
    }
}

extern "C" void kernel_launch(void* const* d_in, const int* in_sizes, int n_in,
                              void* d_out, int out_size, void* d_ws, size_t ws_size,
                              hipStream_t stream)
{
    const float* x     = (const float*)d_in[0];   // [8, 4096] fp32
    const int*   wq    = (const int*)  d_in[1];   // [11008, 4096] int8-in-int32
    const float* scale = (const float*)d_in[2];   // [11008, 1] fp32
    const int*   zp    = (const int*)  d_in[3];   // [11008, 1] int32 (zeros)
    const float* bias  = (const float*)d_in[4];   // [11008] fp32
    float*       out   = (float*)d_out;           // [8, 11008] fp32

    const size_t n_x      = (size_t)TOKENS * IN_FEATURES;          // 32768
    const size_t ws_need  = n_x * sizeof(_Float16) * 2             // xhi + xlo
                          + TOKENS * sizeof(float);                // rowsums

    if (ws_size >= ws_need) {
        _Float16* xhi    = (_Float16*)d_ws;
        _Float16* xlo    = xhi + n_x;
        float*    rowsum = (float*)(xlo + n_x);

        prep_split_kernel<<<dim3(n_x / 256), dim3(256), 0, stream>>>(x, xhi, xlo);
        prep_rowsum_kernel<<<dim3(TOKENS), dim3(256), 0, stream>>>(x, rowsum);
        qlinear_wmma_fast<<<dim3(OUT_FEATURES / 16), dim3(256), 0, stream>>>(
            xhi, xlo, wq, scale, zp, bias, rowsum, out);
    } else {
        qlinear_wmma_fallback<<<dim3(OUT_FEATURES / 16), dim3(256), 0, stream>>>(
            x, wq, scale, zp, bias, out);
    }
}
